// GTNN_5884105195923
// MI455X (gfx1250) — matrix operand order repaired and checked by reading, not compiled, not run on hardware
//
#include <hip/hip_runtime.h>
#include <math.h>

typedef __attribute__((ext_vector_type(16))) _Float16 v16h;
typedef __attribute__((ext_vector_type(8)))  _Float16 v8h;
typedef __attribute__((ext_vector_type(8)))  float    v8f;

#define CDIV(a, b) (((a) + (b) - 1) / (b))
static constexpr float BN_EPS = 1e-5f;

// ---------------------------------------------------------------------------
// small helpers
// ---------------------------------------------------------------------------
__device__ __forceinline__ void atomic_max_f32(float* addr, float val) {
  unsigned int* ua = reinterpret_cast<unsigned int*>(addr);
  unsigned int cur = *ua;
  while (__uint_as_float(cur) < val) {
    unsigned int assumed = cur;
    cur = atomicCAS(ua, assumed, __float_as_uint(val));
    if (cur == assumed) break;
  }
}

__global__ void gtnn_fill(float* __restrict__ p, float v, int n) {
  int t = blockIdx.x * blockDim.x + threadIdx.x;
  if (t < n) p[t] = v;
}

// x stored [ENC, N]; emit zero-padded row-major f16 [N, dinp]
__global__ void gtnn_cvt_xT(const float* __restrict__ x, _Float16* __restrict__ o,
                            int N, int enc, int dinp) {
  int t = blockIdx.x * blockDim.x + threadIdx.x;
  if (t >= N * dinp) return;
  int n = t / dinp, c = t % dinp;
  o[t] = (c < enc) ? (_Float16)x[(size_t)c * N + n] : (_Float16)0.0f;
}

// concat two row-major f32 sources -> zero-padded f16 [N, dinp]
__global__ void gtnn_cvt_cat(const float* __restrict__ a, int da,
                             const float* __restrict__ b, int db,
                             _Float16* __restrict__ o, int N, int dinp) {
  int t = blockIdx.x * blockDim.x + threadIdx.x;
  if (t >= N * dinp) return;
  int n = t / dinp, c = t % dinp;
  float v = 0.0f;
  if (c < da)            v = a[(size_t)n * da + c];
  else if (c < da + db)  v = b[(size_t)n * db + (c - da)];
  o[t] = (_Float16)v;
}

// single weight: f32 [din,dout] -> zero-padded f16 [dinp,doutp]; bias padded
__global__ void gtnn_cvt_w(const float* __restrict__ W, const float* __restrict__ b,
                           _Float16* __restrict__ Wp, float* __restrict__ bp,
                           int din, int dout, int dinp, int doutp) {
  int t = blockIdx.x * blockDim.x + threadIdx.x;
  if (t >= dinp * doutp) return;
  int r = t / doutp, c = t % doutp;
  Wp[t] = (r < din && c < dout) ? (_Float16)W[r * dout + c] : (_Float16)0.0f;
  if (r == 0) bp[c] = (c < dout) ? b[c] : 0.0f;
}

// pack Wq|Wk|Wv|Ws -> zero-padded f16 [dinp, d4p]; biases -> padded f32 [d4p]
__global__ void gtnn_pack_qkvs_w(const float* __restrict__ Wq, const float* __restrict__ Wk,
                                 const float* __restrict__ Wv, const float* __restrict__ Ws,
                                 const float* __restrict__ bq, const float* __restrict__ bk,
                                 const float* __restrict__ bv, const float* __restrict__ bs,
                                 _Float16* __restrict__ Wp, float* __restrict__ bp,
                                 int din, int dh, int dinp, int d4p) {
  int dtot = 4 * dh;
  int t = blockIdx.x * blockDim.x + threadIdx.x;
  if (t >= dinp * d4p) return;
  int r = t / d4p, col = t % d4p;
  float wv = 0.0f, bvv = 0.0f;
  if (r < din && col < dtot) {
    int sel = col / dh, c = col % dh;
    const float* W = (sel == 0) ? Wq : (sel == 1) ? Wk : (sel == 2) ? Wv : Ws;
    wv = W[r * dh + c];
  }
  Wp[t] = (_Float16)wv;
  if (r == 0) {
    if (col < dtot) {
      int sel = col / dh, c = col % dh;
      const float* bb = (sel == 0) ? bq : (sel == 1) ? bk : (sel == 2) ? bv : bs;
      bvv = bb[c];
    }
    bp[col] = bvv;
  }
}

// ---------------------------------------------------------------------------
// WMMA GEMM: C[N,doutp] = A16[N,dinp](f16) @ W16[dinp,doutp](f16) + bias
// All dims pre-padded (dinp % 32 == 0, doutp % 16 == 0, data zero-filled),
// so the k-loop is branch-free: aligned vector loads + v_wmma only.
// One wave per 16x16 output tile. Fragment packing per CDNA5 ISA 7.12.2.
// ---------------------------------------------------------------------------
__global__ void gtnn_wmma_gemm(const _Float16* __restrict__ A,
                               const _Float16* __restrict__ W,
                               const float* __restrict__ bias,
                               float* __restrict__ C,
                               int N, int dinp, int doutp) {
  const int tile_m = blockIdx.x;
  const int n0     = blockIdx.y * 16;
  const int lane   = threadIdx.x;          // full wave, no divergence
  const int m      = lane & 15;
  const int row    = tile_m * 16 + m;
  const int kbase  = (lane < 16) ? 0 : 8;  // 16-bit A 16x32 swizzle

  const _Float16* arow = A + (size_t)row * dinp + kbase;
  const _Float16* wrow = W + (size_t)kbase * doutp + n0 + m;  // B: column n0+m, K rows kbase-swizzled

  v8f acc = {};
  for (int k0 = 0; k0 < dinp; k0 += 32) {
    v8h alo = *(const v8h*)(arow + k0);        // K = k0+kbase .. +7
    v8h ahi = *(const v8h*)(arow + k0 + 16);   // K = k0+kbase+16 .. +23
    v16h a = __builtin_shufflevector(alo, ahi,
                                     0, 1, 2, 3, 4, 5, 6, 7,
                                     8, 9, 10, 11, 12, 13, 14, 15);
    v16h b;
#pragma unroll
    for (int s = 0; s < 16; ++s) b[s] = wrow[(size_t)(k0 + (s < 8 ? s : s + 8)) * doutp];  // K same as a
    acc = __builtin_amdgcn_wmma_f32_16x16x32_f16(
        /*neg_a=*/false, a, /*neg_b=*/false, b,
        /*c_mod=*/(short)0, acc, /*reuse_a=*/false, /*reuse_b=*/false);
  }

  const int mc = (lane < 16) ? 0 : 8;          // f32 C/D layout
  const int nc = lane & 15;
  const float bv = bias[n0 + nc];
  float* crow = C + (size_t)(tile_m * 16 + mc) * doutp + n0 + nc;
#pragma unroll
  for (int r = 0; r < 8; ++r) crow[(size_t)r * doutp] = acc[r] + bv;
}

// ---------------------------------------------------------------------------
// edge-phase kernels (segment softmax + weighted aggregation)
// QKVS row layout (stride = strd): [q(dout) | k(dout) | v(dout) | s(dout) | pad]
// ---------------------------------------------------------------------------
__global__ void gtnn_edge_logit_max(const float* __restrict__ qkvs,
                                    const int* __restrict__ src, const int* __restrict__ dst,
                                    float* __restrict__ logit, float* __restrict__ mbuf,
                                    int E, int H, int Cc, int dout, int strd, float scale) {
  int t = blockIdx.x * blockDim.x + threadIdx.x;
  if (t >= E * H) return;
  int e = t / H, h = t % H;
  int s = src[e], d = dst[e];
  const float* qp = qkvs + (size_t)d * strd + h * Cc;          // q of dst
  const float* kp = qkvs + (size_t)s * strd + dout + h * Cc;   // k of src
  float acc = 0.0f;
  for (int c = 0; c < Cc; ++c) acc += qp[c] * kp[c];
  acc *= scale;
  logit[t] = acc;
  atomic_max_f32(&mbuf[(size_t)d * H + h], acc);
}

__global__ void gtnn_fix_m(float* __restrict__ m, int n) {
  int t = blockIdx.x * blockDim.x + threadIdx.x;
  if (t >= n) return;
  float v = m[t];
  if (!isfinite(v)) m[t] = 0.0f;             // nodes with no incoming edges
}

__global__ void gtnn_edge_exp(const int* __restrict__ dst, float* __restrict__ logit,
                              const float* __restrict__ m, float* __restrict__ den,
                              int E, int H) {
  int t = blockIdx.x * blockDim.x + threadIdx.x;
  if (t >= E * H) return;
  int e = t / H, h = t % H;
  int d = dst[e];
  float v = __expf(logit[t] - m[(size_t)d * H + h]);
  logit[t] = v;
  atomicAdd(&den[(size_t)d * H + h], v);
}

__global__ void gtnn_edge_agg(const float* __restrict__ qkvs,
                              const int* __restrict__ src, const int* __restrict__ dst,
                              const float* __restrict__ expv, const float* __restrict__ den,
                              float* __restrict__ agg, int E, int H, int Cc,
                              int dout, int strd) {
  int t = blockIdx.x * blockDim.x + threadIdx.x;
  if (t >= E * dout) return;
  int e = t / dout, c = t % dout;
  int h = c / Cc;
  int s = src[e], d = dst[e];
  float alpha = expv[(size_t)e * H + h] / fmaxf(den[(size_t)d * H + h], 1e-16f);
  float vv = qkvs[(size_t)s * strd + 2 * dout + c];            // v of src
  atomicAdd(&agg[(size_t)d * dout + c], alpha * vv);
}

// y = agg + skip (in place into agg); accumulate per-channel sum / sumsq
__global__ void gtnn_skip_sum(float* __restrict__ agg, const float* __restrict__ qkvs,
                              float* __restrict__ stats, int N, int dout, int strd) {
  int t = blockIdx.x * blockDim.x + threadIdx.x;
  if (t >= N * dout) return;
  int n = t / dout, c = t % dout;
  float v = agg[t] + qkvs[(size_t)n * strd + 3 * dout + c];
  agg[t] = v;
  atomicAdd(&stats[c], v);
  atomicAdd(&stats[dout + c], v * v);
}

__global__ void gtnn_bn_relu(const float* __restrict__ y, const float* __restrict__ stats,
                             const float* __restrict__ g, const float* __restrict__ b,
                             float* __restrict__ out, int N, int dout) {
  int t = blockIdx.x * blockDim.x + threadIdx.x;
  if (t >= N * dout) return;
  int c = t % dout;
  float inv_n = 1.0f / (float)N;
  float mu  = stats[c] * inv_n;
  float var = stats[dout + c] * inv_n - mu * mu;
  float v = g[c] * (y[t] - mu) * rsqrtf(var + BN_EPS) + b[c];
  out[t] = fmaxf(v, 0.0f);
}

// conv5 epilogue: out = softmax(agg + skip) over 2 classes
__global__ void gtnn_final_softmax(const float* __restrict__ agg, const float* __restrict__ qkvs,
                                   float* __restrict__ out, int N, int strd) {
  int n = blockIdx.x * blockDim.x + threadIdx.x;
  if (n >= N) return;
  float a0 = agg[(size_t)n * 2 + 0] + qkvs[(size_t)n * strd + 6 + 0];
  float a1 = agg[(size_t)n * 2 + 1] + qkvs[(size_t)n * strd + 6 + 1];
  float mx = fmaxf(a0, a1);
  float e0 = __expf(a0 - mx), e1 = __expf(a1 - mx);
  float inv = 1.0f / (e0 + e1);
  out[(size_t)n * 2 + 0] = e0 * inv;
  out[(size_t)n * 2 + 1] = e1 * inv;
}

// ---------------------------------------------------------------------------
// host orchestration
// ---------------------------------------------------------------------------
static inline int pad_to(int x, int p) { return (x + p - 1) & ~(p - 1); }

extern "C" void kernel_launch(void* const* d_in, const int* in_sizes, int n_in,
                              void* d_out, int out_size, void* d_ws, size_t ws_size,
                              hipStream_t stream) {
  const int N = in_sizes[0] / 16;          // x is [16, N]
  const int E = in_sizes[1] / 2;           // edge_index is [2, E]

  const float* x  = (const float*)d_in[0];
  const int* src  = (const int*)d_in[1];
  const int* dst  = src + E;

  // param leaf order: fc{W,b}, conv1..5{Wq,bq,Wk,bk,Wv,bv,Ws,bs}, bn1..4{g,b}
  const float* fcW = (const float*)d_in[2];
  const float* fcB = (const float*)d_in[3];
  const int conv_base[5] = {4, 12, 20, 28, 36};
  const int bn_base[4]   = {44, 46, 48, 50};

  // --- workspace partition -------------------------------------------------
  char* wsp = (char*)d_ws;
  auto take = [&](size_t bytes) -> void* {
    void* p = (void*)wsp;
    wsp += (bytes + 255) & ~(size_t)255;
    return p;
  };
  float*    h    = (float*)take((size_t)N * 32 * 4);
  float*    x1   = (float*)take((size_t)N * 32 * 4);
  float*    x2   = (float*)take((size_t)N * 32 * 4);
  float*    x3   = (float*)take((size_t)N * 32 * 4);
  float*    x4   = (float*)take((size_t)N * 32 * 4);
  _Float16* A16  = (_Float16*)take((size_t)N * 64 * 2);
  float*    QKVS = (float*)take((size_t)N * 128 * 4);
  float*    LOGB = (float*)take((size_t)E * 4 * 4);
  float*    M    = (float*)take((size_t)N * 4 * 4);
  float*    DEN  = (float*)take((size_t)N * 4 * 4);
  float*    AGG  = (float*)take((size_t)N * 32 * 4);
  _Float16* WP   = (_Float16*)take((size_t)64 * 128 * 2);
  float*    BP   = (float*)take((size_t)128 * 4);
  float*    STATS= (float*)take((size_t)64 * 4);

  const int tiles_m = CDIV(N, 16);

  // --- stage 0: h = x^T @ Wfc + bfc  (din 16 -> padded 32) -----------------
  {
    const int dinp = 32, doutp = 32;
    gtnn_cvt_xT<<<CDIV(N * dinp, 256), 256, 0, stream>>>(x, A16, N, 16, dinp);
    gtnn_cvt_w<<<CDIV(dinp * doutp, 256), 256, 0, stream>>>(
        fcW, fcB, WP, BP, 16, 32, dinp, doutp);
    dim3 g(tiles_m, doutp / 16);
    gtnn_wmma_gemm<<<g, 32, 0, stream>>>(A16, WP, BP, h, N, dinp, doutp);
  }

  // --- generic TransformerConv layer --------------------------------------
  auto run_tconv = [&](const float* ina, int da, const float* inb, int db,
                       int pb, int H, int Cc,
                       const float* bn_g, const float* bn_b, float* xout) {
    const int din  = da + db;
    const int dout = H * Cc;
    const int dinp = pad_to(din, 32);       // K padded to WMMA depth
    const int d4   = 4 * dout;
    const int d4p  = pad_to(d4, 16);        // N padded to tile width

    gtnn_cvt_cat<<<CDIV(N * dinp, 256), 256, 0, stream>>>(ina, da, inb, db, A16, N, dinp);
    gtnn_pack_qkvs_w<<<CDIV(dinp * d4p, 256), 256, 0, stream>>>(
        (const float*)d_in[pb + 0], (const float*)d_in[pb + 2],
        (const float*)d_in[pb + 4], (const float*)d_in[pb + 6],
        (const float*)d_in[pb + 1], (const float*)d_in[pb + 3],
        (const float*)d_in[pb + 5], (const float*)d_in[pb + 7],
        WP, BP, din, dout, dinp, d4p);
    {
      dim3 g(tiles_m, d4p / 16);
      gtnn_wmma_gemm<<<g, 32, 0, stream>>>(A16, WP, BP, QKVS, N, dinp, d4p);
    }

    gtnn_fill<<<CDIV(N * 4, 256), 256, 0, stream>>>(M, -INFINITY, N * 4);
    gtnn_fill<<<CDIV(N * 4, 256), 256, 0, stream>>>(DEN, 0.0f, N * 4);
    gtnn_fill<<<CDIV(N * dout, 256), 256, 0, stream>>>(AGG, 0.0f, N * dout);

    const float scale = 1.0f / sqrtf((float)Cc);
    gtnn_edge_logit_max<<<CDIV(E * H, 256), 256, 0, stream>>>(
        QKVS, src, dst, LOGB, M, E, H, Cc, dout, d4p, scale);
    gtnn_fix_m<<<CDIV(N * H, 256), 256, 0, stream>>>(M, N * H);
    gtnn_edge_exp<<<CDIV(E * H, 256), 256, 0, stream>>>(dst, LOGB, M, DEN, E, H);
    gtnn_edge_agg<<<CDIV(E * dout, 256), 256, 0, stream>>>(
        QKVS, src, dst, LOGB, DEN, AGG, E, H, Cc, dout, d4p);

    if (bn_g) {
      gtnn_fill<<<1, 64, 0, stream>>>(STATS, 0.0f, 64);
      gtnn_skip_sum<<<CDIV(N * dout, 256), 256, 0, stream>>>(AGG, QKVS, STATS, N, dout, d4p);
      gtnn_bn_relu<<<CDIV(N * dout, 256), 256, 0, stream>>>(
          AGG, STATS, bn_g, bn_b, xout, N, dout);
    } else {
      gtnn_final_softmax<<<CDIV(N, 256), 256, 0, stream>>>(
          AGG, QKVS, (float*)d_out, N, d4p);
    }
  };

  // conv1: h[N,32], H=4, C=8 -> x1
  run_tconv(h, 32, nullptr, 0, conv_base[0], 4, 8,
            (const float*)d_in[bn_base[0]], (const float*)d_in[bn_base[0] + 1], x1);
  // conv2: x1 -> x2
  run_tconv(x1, 32, nullptr, 0, conv_base[1], 4, 8,
            (const float*)d_in[bn_base[1]], (const float*)d_in[bn_base[1] + 1], x2);
  // conv3: x2, H=1, C=32 -> x3
  run_tconv(x2, 32, nullptr, 0, conv_base[2], 1, 32,
            (const float*)d_in[bn_base[2]], (const float*)d_in[bn_base[2] + 1], x3);
  // conv4: cat(x1,x3)[N,64], H=1, C=32 -> x4
  run_tconv(x1, 32, x3, 32, conv_base[3], 1, 32,
            (const float*)d_in[bn_base[3]], (const float*)d_in[bn_base[3] + 1], x4);
  // conv5: cat(x2,x4)[N,64], H=1, C=2 -> softmax -> d_out
  run_tconv(x2, 32, x4, 32, conv_base[4], 1, 2, nullptr, nullptr, nullptr);
}
